// Net_45896020525235
// MI455X (gfx1250) — matrix operand order repaired and checked by reading, not compiled, not run on hardware
//
#include <hip/hip_runtime.h>
#include <hip/hip_bf16.h>

// ---------------------------------------------------------------------------
// CDNA5 (gfx1250, wave32) implementation.
// All dense GEMMs run on v_wmma_f32_16x16x32_bf16 (fp32 accumulate).
// B operands (weights) are pre-swizzled into WMMA fragment order so each
// lane's fragment is a single contiguous 32B load (2x global_load_b128).
// ---------------------------------------------------------------------------

typedef __attribute__((ext_vector_type(16))) __bf16 v16bf;
typedef __attribute__((ext_vector_type(8)))  __bf16 v8bf;
typedef __attribute__((ext_vector_type(8)))  float  v8f;

#define HID 128
#define FEAT 24
#define NCLUS 64

// ---------------------------------------------------------------------------
// Pack (and convert f32->bf16) a row-major weight B[K,N] into per-fragment
// order: dst[((tn*KT + kt)*32 + lane)*16 + e] = B[kt*32 + (e>=8)*16 + (lane>=16)*8 + e%8,
// tn*16 + lane%16].  Rows k >= srcK are zero (K padding for Wg1).
// ---------------------------------------------------------------------------
__global__ void pack_b_kernel(const float* __restrict__ src, __bf16* __restrict__ dst,
                              int N, int K, int srcK) {
    const int KT = K >> 5;
    long long idx = (long long)blockIdx.x * blockDim.x + threadIdx.x;
    const long long tot = (long long)N * K;
    if (idx >= tot) return;
    const int e    = (int)(idx & 15);
    const int lane = (int)((idx >> 4) & 31);
    const int blk  = (int)(idx >> 9);          // tn*KT + kt
    const int kt   = blk % KT;
    const int tn   = blk / KT;
    const int k = kt * 32 + ((e & 8) ? 16 : 0) + ((lane >> 4) ? 8 : 0) + (e & 7);
    const int n = tn * 16 + (lane & 15);
    dst[idx] = (__bf16)((k < srcK) ? src[(size_t)k * N + n] : 0.f);
}

// ---------------------------------------------------------------------------
// bf16 WMMA GEMM: C[M, NT*16] (f32) = A[M,K](bf16, lda==K) * Bp (packed bf16)
// One wave computes a 16 x (NT*16) strip: A fragment loaded once per k-step,
// reused across NT accumulator tiles. 4 strips per 128-thread block.
// Fragment layouts per CDNA5 ISA §7.12.2 (wave32).
// ---------------------------------------------------------------------------
template<int NT>
__global__ __launch_bounds__(128) void gemm_bf16_wmma(
    const __bf16* __restrict__ A, const __bf16* __restrict__ Bp,
    float* __restrict__ C, __bf16* __restrict__ Cbf,
    int Mtiles, int K, int do_relu)
{
    const int N    = NT * 16;
    const int KT   = K >> 5;
    const int lane = threadIdx.x & 31;
    const int tile = blockIdx.x * 4 + (threadIdx.x >> 5);
    if (tile >= Mtiles) return;                 // wave-uniform: EXEC stays all-1s
    const int row0 = tile << 4;
    const int hi   = lane >> 4;                 // 0: lanes 0-15, 1: lanes 16-31

    v8f acc[NT];
#pragma unroll
    for (int t = 0; t < NT; ++t) { v8f z = {}; acc[t] = z; }

    const __bf16* ap0 = A + (size_t)(row0 + (lane & 15)) * K + (hi ? 8 : 0);
    for (int kt = 0; kt < KT; ++kt) {
        // A fragment 16x32: two contiguous 16B chunks per lane (-> b128 loads)
        const v8bf lo = *(const v8bf*)(ap0 + kt * 32);
        const v8bf hh = *(const v8bf*)(ap0 + kt * 32 + 16);
        v16bf a;
#pragma unroll
        for (int e = 0; e < 8; ++e) { a[e] = lo[e]; a[8 + e] = hh[e]; }
#pragma unroll
        for (int t = 0; t < NT; ++t) {
            // packed B fragment: contiguous 32B per lane, 1KB per wave
            const v16bf b = *(const v16bf*)(Bp + ((size_t)(t * KT + kt) * 32 + lane) * 16);
            acc[t] = __builtin_amdgcn_wmma_f32_16x16x32_bf16(
                /*neg_a=*/false, a, /*neg_b=*/false, b,
                /*c_mod=*/(short)0, acc[t], /*reuse_a=*/false, /*reuse_b=*/false);
        }
    }

    const int mbase = row0 + (hi ? 8 : 0);
#pragma unroll
    for (int t = 0; t < NT; ++t) {
        const int ncol = t * 16 + (lane & 15);
#pragma unroll
        for (int r = 0; r < 8; ++r) {
            float v = acc[t][r];
            if (do_relu) v = v > 0.f ? v : 0.f;
            const size_t idx = (size_t)(mbase + r) * N + ncol;
            if (C)   C[idx]   = v;
            if (Cbf) Cbf[idx] = (__bf16)v;
        }
    }
}

// ---------------------------------------------------------------------------
// Utility kernels
// ---------------------------------------------------------------------------
__global__ void zero_kernel(float* __restrict__ p, long long n) {
    long long i = (long long)blockIdx.x * blockDim.x + threadIdx.x;
    const long long stride = (long long)gridDim.x * blockDim.x;
    for (; i < n; i += stride) p[i] = 0.f;
}

// f32 -> bf16 with zero padding (dst [drows,dcols], src [srows,scols])
__global__ void cvt_bf16_kernel(const float* __restrict__ src, __bf16* __restrict__ dst,
                                int srows, int scols, int drows, int dcols) {
    long long i = (long long)blockIdx.x * blockDim.x + threadIdx.x;
    const long long tot = (long long)drows * dcols;
    if (i >= tot) return;
    const int r = (int)(i / dcols), c = (int)(i % dcols);
    float v = (r < srows && c < scols) ? src[(size_t)r * scols + c] : 0.f;
    dst[i] = (__bf16)v;
}

__global__ void deg_kernel(const int* __restrict__ dst, float* __restrict__ deg, int E) {
    int e = blockIdx.x * blockDim.x + threadIdx.x;
    if (e < E) atomicAdd(&deg[dst[e]], 1.0f);
}

__global__ void dinv_kernel(float* __restrict__ deg, int n) {
    int i = blockIdx.x * blockDim.x + threadIdx.x;
    if (i < n) deg[i] = rsqrtf(deg[i] + 1.0f);   // +1 accounts for the self loop
}

// Symmetric-normalized scatter-add over real edges (self loops handled in
// the finalize kernels). One block per edge, one thread per feature.
__global__ void prop_kernel(const float* __restrict__ H, const int* __restrict__ src,
                            const int* __restrict__ dst, const float* __restrict__ dinv,
                            float* __restrict__ out, int F) {
    const int e = blockIdx.x;
    const int f = threadIdx.x;
    const int s = src[e], d = dst[e];
    const float w = dinv[s] * dinv[d];
    atomicAdd(&out[(size_t)d * F + f], H[(size_t)s * F + f] * w);
}

// layer1 finalize: prop + self-loop + bias, relu, emit bf16 operand
__global__ void finalize1_kernel(const float* __restrict__ prop, const float* __restrict__ h,
                                 const float* __restrict__ dinv, const float* __restrict__ bias,
                                 __bf16* __restrict__ outb, long long total, int F) {
    long long i = (long long)blockIdx.x * blockDim.x + threadIdx.x;
    if (i >= total) return;
    const int node = (int)(i / F), f = (int)(i % F);
    const float dv = dinv[node];
    float v = prop[i] + h[i] * dv * dv + bias[f];
    outb[i] = (__bf16)(v > 0.f ? v : 0.f);
}

// layer2 finalize: prop + self-loop + bias (fp32 in place, no relu)
__global__ void finalize2_kernel(float* __restrict__ p, const float* __restrict__ h,
                                 const float* __restrict__ dinv, const float* __restrict__ bias,
                                 long long total, int F) {
    long long i = (long long)blockIdx.x * blockDim.x + threadIdx.x;
    if (i >= total) return;
    const int node = (int)(i / F), f = (int)(i % F);
    const float dv = dinv[node];
    p[i] = p[i] + h[i] * dv * dv + bias[f];
}

// Fused: row-wise log_softmax (in place, fp32) + bf16 copy + column sum
// (for graph_vec) + cluster segment-sum. One 128-thread block per node.
__global__ __launch_bounds__(128) void logsoftmax_kernel(
    float* __restrict__ h, __bf16* __restrict__ nvb, const int* __restrict__ clusters,
    float* __restrict__ colsum, float* __restrict__ cluster_vec)
{
    __shared__ float s[HID];
    const int node = blockIdx.x;
    const int f = threadIdx.x;
    const float v = h[(size_t)node * HID + f];
    s[f] = v; __syncthreads();
    for (int off = 64; off > 0; off >>= 1) {
        if (f < off) s[f] = fmaxf(s[f], s[f + off]);
        __syncthreads();
    }
    const float mx = s[0]; __syncthreads();
    s[f] = __expf(v - mx); __syncthreads();
    for (int off = 64; off > 0; off >>= 1) {
        if (f < off) s[f] += s[f + off];
        __syncthreads();
    }
    const float nv = v - mx - __logf(s[0]);
    h[(size_t)node * HID + f]  = nv;
    nvb[(size_t)node * HID + f] = (__bf16)nv;
    atomicAdd(&colsum[f], nv);
    const int cid = clusters[2 * node + 1];
    atomicAdd(&cluster_vec[(size_t)cid * HID + f], nv);
}

// graph_vec = colsum @ W2 ; gconst = sum(relu(graph_vec) * W1[0:128])
__global__ __launch_bounds__(128) void graph_kernel(
    const float* __restrict__ colsum, const float* __restrict__ W2,
    const float* __restrict__ W1, float* __restrict__ gconst)
{
    __shared__ float s[HID];
    const int j = threadIdx.x;
    float acc = 0.f;
    #pragma unroll 4
    for (int k = 0; k < HID; ++k) acc += colsum[k] * W2[k * HID + j];
    acc = acc > 0.f ? acc : 0.f;
    s[j] = acc * W1[j];
    __syncthreads();
    for (int off = 64; off > 0; off >>= 1) {
        if (j < off) s[j] += s[j + off];
        __syncthreads();
    }
    if (j == 0) gconst[0] = s[0];
}

// out[m] = gconst + sum_f relu(rc5a[Bi]+rn4b[Bj]+rc5c[Bk]+rn4d[Bl])[f] * W1[128+f]
// One wave per candidate row; shuffle reduction.
__global__ __launch_bounds__(256) void final_kernel(
    const int* __restrict__ Bm, const float* __restrict__ rc5a,
    const float* __restrict__ rn4b, const float* __restrict__ rc5c,
    const float* __restrict__ rn4d, const float* __restrict__ w1b,
    const float* __restrict__ gconst, float* __restrict__ out, int M)
{
    const int lane = threadIdx.x & 31;
    const int w    = threadIdx.x >> 5;
    const int m    = blockIdx.x * 8 + w;
    if (m >= M) return;
    const int bi = Bm[4 * m + 0], bj = Bm[4 * m + 1];
    const int bk = Bm[4 * m + 2], bl = Bm[4 * m + 3];
    const float* pa = rc5a + (size_t)bi * HID;
    const float* pb = rn4b + (size_t)bj * HID;
    const float* pc = rc5c + (size_t)bk * HID;
    const float* pd = rn4d + (size_t)bl * HID;
    float acc = 0.f;
    #pragma unroll
    for (int c = 0; c < 4; ++c) {
        const int f = lane + 32 * c;
        const float t = pa[f] + pb[f] + pc[f] + pd[f];
        acc += (t > 0.f ? t : 0.f) * w1b[f];
    }
    #pragma unroll
    for (int off = 16; off > 0; off >>= 1) acc += __shfl_xor(acc, off, 32);
    if (lane == 0) out[m] = acc + gconst[0];
}

// ---------------------------------------------------------------------------
// Host orchestration
// ---------------------------------------------------------------------------
extern "C" void kernel_launch(void* const* d_in, const int* in_sizes, int n_in,
                              void* d_out, int out_size, void* d_ws, size_t ws_size,
                              hipStream_t stream) {
    (void)n_in; (void)out_size; (void)ws_size;
    const int N = in_sizes[0] / FEAT;     // 50000 (divisible by 16)
    const int E = in_sizes[1] / 2;        // 800000
    const int M = in_sizes[2] / 4;        // 100000

    const float* x      = (const float*)d_in[0];
    const int*   ei     = (const int*)  d_in[1];
    const int*   src    = ei;
    const int*   dst    = ei + E;
    const int*   Bm     = (const int*)  d_in[2];
    const int*   clus   = (const int*)  d_in[3];
    const float* Wg1    = (const float*)d_in[4];
    const float* bg1    = (const float*)d_in[5];
    const float* Wg2    = (const float*)d_in[6];
    const float* bg2    = (const float*)d_in[7];
    const float* W1     = (const float*)d_in[8];   // [256,1]
    const float* W2     = (const float*)d_in[9];   // [128,128]
    const float* W3     = (const float*)d_in[10];  // [512,128]
    const float* W4     = (const float*)d_in[11];
    const float* W5     = (const float*)d_in[12];
    float* out = (float*)d_out;

    // ---- workspace layout (explicit reuse) ----
    char* ws = (char*)d_ws;
    size_t o = 0;
    auto take = [&](size_t bytes) -> char* {
        char* p = ws + o; o += (bytes + 255) & ~(size_t)255; return p;
    };
    float*  colsum  = (float*)take(HID * sizeof(float));
    float*  gconst  = (float*)take(sizeof(float));
    float*  degv    = (float*)take((size_t)N * sizeof(float));          // deg -> dinv
    float*  cvec    = (float*)take((size_t)NCLUS * HID * sizeof(float));
    __bf16* cvb     = (__bf16*)take((size_t)NCLUS * HID * 2);
    __bf16* c5rb    = (__bf16*)take((size_t)NCLUS * HID * 2);
    float*  rc5a    = (float*)take((size_t)NCLUS * HID * sizeof(float));
    float*  rc5c    = (float*)take((size_t)NCLUS * HID * sizeof(float));
    __bf16* wg1p    = (__bf16*)take((size_t)32 * 96 * 2);               // packed, K 24->32
    __bf16* wg2p    = (__bf16*)take((size_t)96 * HID * 2);
    __bf16* w4p     = (__bf16*)take((size_t)HID * HID * 2);
    __bf16* w5p     = (__bf16*)take((size_t)HID * HID * 2);
    __bf16* w3p[4];
    for (int q = 0; q < 4; ++q) w3p[q] = (__bf16*)take((size_t)HID * HID * 2);
    __bf16* xb      = (__bf16*)take((size_t)N * 32 * 2);                // x padded bf16
    __bf16* h1b     = (__bf16*)take((size_t)N * 96 * 2);                // relu(layer1) bf16
    // R1 (38.4 MB): h1 + p1  -> later xw2 -> later nvb + n4rb
    char*   R1      = take((size_t)N * 96 * 2 * sizeof(float));
    float*  h1      = (float*)R1;                                       // x @ Wg1
    float*  p1      = (float*)(R1 + (size_t)N * 96 * sizeof(float));    // propagated
    float*  xw2     = (float*)R1;                                       // h1b @ Wg2
    __bf16* nvb     = (__bf16*)R1;                                      // nodes_vec bf16
    __bf16* n4rb    = (__bf16*)(R1 + (size_t)N * HID * 2);              // relu(n4) bf16
    // R2 (25.6 MB): p2 / nodes_vec f32  -> later rn4b
    char*   R2      = take((size_t)N * HID * sizeof(float));
    float*  p2      = (float*)R2;
    float*  rn4b    = (float*)R2;
    // R3 (25.6 MB): rn4d
    float*  rn4d    = (float*)take((size_t)N * HID * sizeof(float));

    const int ZB = 2048, TB = 256;
    // ---- zero accumulators ----
    zero_kernel<<<ZB, TB, 0, stream>>>(degv, (long long)N);
    zero_kernel<<<ZB, TB, 0, stream>>>(p1, (long long)N * 96);
    zero_kernel<<<ZB, TB, 0, stream>>>(colsum, HID);
    zero_kernel<<<ZB, TB, 0, stream>>>(cvec, (long long)NCLUS * HID);

    // ---- weight packing (f32 -> bf16, WMMA fragment order) ----
    auto packb = [&](const float* s, __bf16* d, int Nn, int Kk, int srcK) {
        long long tot = (long long)Nn * Kk;
        pack_b_kernel<<<(unsigned)((tot + TB - 1) / TB), TB, 0, stream>>>(s, d, Nn, Kk, srcK);
    };
    packb(Wg1, wg1p, 96, 32, FEAT);
    packb(Wg2, wg2p, HID, 96, 96);
    packb(W4,  w4p,  HID, HID, HID);
    packb(W5,  w5p,  HID, HID, HID);
    for (int q = 0; q < 4; ++q)
        packb(W3 + (size_t)q * HID * HID, w3p[q], HID, HID, HID);

    // ---- A-side conversions ----
    auto cvt = [&](const float* s, __bf16* d, int sr, int sc, int dr, int dc) {
        long long tot = (long long)dr * dc;
        cvt_bf16_kernel<<<(unsigned)((tot + TB - 1) / TB), TB, 0, stream>>>(s, d, sr, sc, dr, dc);
    };
    cvt(x, xb, N, FEAT, N, 32);

    // ---- degree / normalization ----
    deg_kernel<<<(E + TB - 1) / TB, TB, 0, stream>>>(dst, degv, E);
    dinv_kernel<<<(N + TB - 1) / TB, TB, 0, stream>>>(degv, N);

    const int MT  = N / 16;                   // node row-tiles
    const int GMT = (MT + 3) / 4;
    const int CT  = NCLUS / 16;               // cluster row-tiles

    // ---- GCN layer 1: h1 = x @ Wg1 ----
    gemm_bf16_wmma<6><<<GMT, 128, 0, stream>>>(xb, wg1p, h1, nullptr, MT, 32, 0);
    prop_kernel<<<E, 96, 0, stream>>>(h1, src, dst, degv, p1, 96);
    {
        long long tot = (long long)N * 96;
        finalize1_kernel<<<(unsigned)((tot + TB - 1) / TB), TB, 0, stream>>>(
            p1, h1, degv, bg1, h1b, tot, 96);
    }

    // ---- GCN layer 2: xw2 = h1 @ Wg2 ----
    gemm_bf16_wmma<8><<<GMT, 128, 0, stream>>>(h1b, wg2p, xw2, nullptr, MT, 96, 0);
    zero_kernel<<<ZB, TB, 0, stream>>>(p2, (long long)N * HID);
    prop_kernel<<<E, HID, 0, stream>>>(xw2, src, dst, degv, p2, HID);
    {
        long long tot = (long long)N * HID;
        finalize2_kernel<<<(unsigned)((tot + TB - 1) / TB), TB, 0, stream>>>(
            p2, xw2, degv, bg2, tot, HID);
    }

    // ---- log_softmax + pooling (fused) ----
    logsoftmax_kernel<<<N, HID, 0, stream>>>(p2, nvb, clus, colsum, cvec);

    // ---- graph vector scalar contribution ----
    graph_kernel<<<1, HID, 0, stream>>>(colsum, W2, W1, gconst);

    // ---- cluster branch: c5 = relu(cvec @ W5); rc5a/rc5c = c5r @ W3 blocks ----
    cvt(cvec, cvb, NCLUS, HID, NCLUS, HID);
    gemm_bf16_wmma<8><<<1, 128, 0, stream>>>(cvb,  w5p,    nullptr, c5rb, CT, HID, 1);
    gemm_bf16_wmma<8><<<1, 128, 0, stream>>>(c5rb, w3p[0], rc5a, nullptr, CT, HID, 0);
    gemm_bf16_wmma<8><<<1, 128, 0, stream>>>(c5rb, w3p[2], rc5c, nullptr, CT, HID, 0);

    // ---- node branch: n4r = relu(nv @ W4); rn4b/rn4d = n4r @ W3 blocks ----
    gemm_bf16_wmma<8><<<GMT, 128, 0, stream>>>(nvb,  w4p,    nullptr, n4rb, MT, HID, 1);
    gemm_bf16_wmma<8><<<GMT, 128, 0, stream>>>(n4rb, w3p[1], rn4b, nullptr, MT, HID, 0);
    gemm_bf16_wmma<8><<<GMT, 128, 0, stream>>>(n4rb, w3p[3], rn4d, nullptr, MT, HID, 0);

    // ---- candidate scoring ----
    final_kernel<<<(M + 7) / 8, 256, 0, stream>>>(
        Bm, rc5a, rn4b, rc5c, rn4d, W1 + HID, gconst, out, M);
}
